// MultiHeadAttention_79474074845430
// MI455X (gfx1250) — matrix operand order repaired and checked by reading, not compiled, not run on hardware
//
#include <hip/hip_runtime.h>

#define S_LEN 2048
#define DM    1024
#define HNUM  16
#define DKK   64
#define SC_STRIDE 2052   // 32x2048 fp32 score tile, +4 skew: row-pair stores hit disjoint banks

typedef __attribute__((ext_vector_type(16))) __bf16 v16bf;
typedef __attribute__((ext_vector_type(8)))  __bf16 v8bf;
typedef __attribute__((ext_vector_type(8)))  float  v8f;
typedef __attribute__((ext_vector_type(4)))  float  v4f;

// ---- WMMA fragment loaders ----------------------------------------------
// A fragment (16x32 MxK, bf16): lane lr = row M; per-lane K packing is
// 16 contiguous K values starting at hi*16 (matches B fragment).
// 'p' must point at row_base + k0 + hi*16.  Casts -> v_cvt_pk_bf16_f32.
__device__ __forceinline__ v16bf fragA_f32(const float* p) {
    v4f a0 = *reinterpret_cast<const v4f*>(p);
    v4f a1 = *reinterpret_cast<const v4f*>(p + 4);
    v4f b0 = *reinterpret_cast<const v4f*>(p + 8);
    v4f b1 = *reinterpret_cast<const v4f*>(p + 12);
    v16bf f;
#pragma unroll
    for (int i = 0; i < 4; ++i) {
        f[i]      = (__bf16)a0[i];
        f[4 + i]  = (__bf16)a1[i];
        f[8 + i]  = (__bf16)b0[i];
        f[12 + i] = (__bf16)b1[i];
    }
    return f;
}

__device__ __forceinline__ v16bf fragA_bf16(const __bf16* p) {
    v8bf c0 = *reinterpret_cast<const v8bf*>(p);
    v8bf c1 = *reinterpret_cast<const v8bf*>(p + 8);
    return __builtin_shufflevector(c0, c1, 0, 1, 2, 3, 4, 5, 6, 7,
                                   8, 9, 10, 11, 12, 13, 14, 15);
}

// B fragment (32x16 KxN, bf16): lane lr = column N (== a row of the source
// matrix, since B = src^T), 16 contiguous K values starting at hi*16.
// 'p' must point at srcrow_base + k0 + hi*16.
__device__ __forceinline__ v16bf fragB_f32(const float* p) {
    v16bf f;
#pragma unroll
    for (int j = 0; j < 4; ++j) {
        v4f a = *reinterpret_cast<const v4f*>(p + 4 * j);
#pragma unroll
        for (int i = 0; i < 4; ++i) f[4 * j + i] = (__bf16)a[i];
    }
    return f;
}

__device__ __forceinline__ v16bf fragB_bf16(const __bf16* p) {
    v8bf c0 = *reinterpret_cast<const v8bf*>(p);
    v8bf c1 = *reinterpret_cast<const v8bf*>(p + 8);
    return __builtin_shufflevector(c0, c1, 0, 1, 2, 3, 4, 5, 6, 7,
                                   8, 9, 10, 11, 12, 13, 14, 15);
}

#define WMMA_BF16(A, B, C) \
    __builtin_amdgcn_wmma_f32_16x16x32_bf16(false, (A), false, (B), (short)0, (C), false, false)

// -------------------------------------------------------------------------
// Projection GEMM: out[4096x1024] = A[4096x1024] * W^T + bias
// mode 0: write bf16 [B,H,S,dk]   (q, k)
// mode 1: write bf16 [B,H,dk,S]   (v transposed)
// mode 2: write fp32 [B,S,D]      (final output projection)
// One wave computes a 32(M) x 64(N) tile (B fragments reused twice).
__global__ __launch_bounds__(256) void proj_gemm_kernel(
    const float* __restrict__ A, const float* __restrict__ W,
    const float* __restrict__ bias, __bf16* __restrict__ outB,
    float* __restrict__ outF, int mode)
{
    const int lane = threadIdx.x & 31;
    const int wid  = threadIdx.x >> 5;
    const int lr = lane & 15, hi = lane >> 4;
    const int mTile = blockIdx.x * 8 + wid;   // 0..127
    const int nTile = blockIdx.y;             // 0..15
    const int m0 = mTile * 32, n0 = nTile * 64;

    v8f acc[2][4] = {};
    const float* arow0 = A + (size_t)(m0 + lr) * DM + hi * 16;
    const float* arow1 = arow0 + (size_t)16 * DM;
#pragma unroll 2
    for (int kc = 0; kc < DM; kc += 32) {
        v16bf af0 = fragA_f32(arow0 + kc);
        v16bf af1 = fragA_f32(arow1 + kc);
#pragma unroll
        for (int n = 0; n < 4; ++n) {
            const float* wrow = W + (size_t)(n0 + n * 16 + lr) * DM + kc + hi * 16;
            v16bf bf = fragB_f32(wrow);
            acc[0][n] = WMMA_BF16(af0, bf, acc[0][n]);
            acc[1][n] = WMMA_BF16(af1, bf, acc[1][n]);
        }
    }
#pragma unroll
    for (int n = 0; n < 4; ++n) {
        const int gn = n0 + n * 16 + lr;
        const float bval = bias[gn];
        const int h = gn >> 6, d = gn & (DKK - 1);
#pragma unroll
        for (int t = 0; t < 2; ++t) {
#pragma unroll
            for (int r = 0; r < 8; ++r) {
                const int gm = m0 + t * 16 + r + 8 * hi;
                const int b = gm >> 11, s = gm & (S_LEN - 1);
                const float val = acc[t][n][r] + bval;
                if (mode == 0)
                    outB[((size_t)(b * HNUM + h) * S_LEN + s) * DKK + d] = (__bf16)val;
                else if (mode == 1)
                    outB[((size_t)(b * HNUM + h) * DKK + d) * S_LEN + s] = (__bf16)val;
                else
                    outF[(size_t)gm * DM + gn] = val;
            }
        }
    }
}

// -------------------------------------------------------------------------
// Fused attention: scores -> softmax -> P*V for one (bh, 32-query-row) block.
// LDS holds the 32x2048 fp32 score/prob tile; attn is written to global
// exactly once; P*V partials are reduced deterministically across waves.
__global__ __launch_bounds__(256) void fused_attn_kernel(
    const __bf16* __restrict__ qb, const __bf16* __restrict__ kb,
    const __bf16* __restrict__ vtb, float* __restrict__ attn_out,
    float* __restrict__ ctx)
{
    extern __shared__ float sc[];     // 32 * SC_STRIDE floats (262,656 B)
    const int tid  = threadIdx.x;
    const int lane = tid & 31;
    const int wid  = tid >> 5;        // 0..7: owns k-slice [wid*256, wid*256+256)
    const int lr = lane & 15, hi = lane >> 4;
    const int qTile = blockIdx.x;     // 0..63
    const int bh    = blockIdx.y;     // 0..31
    const int q0 = qTile * 32;
    const int b = bh >> 4, h = bh & 15;
    const int kbase = wid * 256;

    // ---- phase 1: raw scores into LDS (q fragments stay resident) -------
    const __bf16* qrow0 = qb + ((size_t)bh * S_LEN + q0 + lr) * DKK + hi * 16;
    const __bf16* qrow1 = qrow0 + (size_t)16 * DKK;
    v16bf qf00 = fragA_bf16(qrow0);
    v16bf qf01 = fragA_bf16(qrow0 + 32);
    v16bf qf10 = fragA_bf16(qrow1);
    v16bf qf11 = fragA_bf16(qrow1 + 32);

#pragma unroll 4
    for (int t = 0; t < 16; ++t) {
        const int k0 = kbase + t * 16;
        const __bf16* krow = kb + ((size_t)bh * S_LEN + k0 + lr) * DKK + hi * 16;
        v16bf kf0 = fragB_bf16(krow);
        v16bf kf1 = fragB_bf16(krow + 32);
        v8f acc0 = {}, acc1 = {};
        acc0 = WMMA_BF16(qf00, kf0, acc0);
        acc0 = WMMA_BF16(qf01, kf1, acc0);
        acc1 = WMMA_BF16(qf10, kf0, acc1);
        acc1 = WMMA_BF16(qf11, kf1, acc1);
#pragma unroll
        for (int r = 0; r < 8; ++r) {
            sc[(r + 8 * hi) * SC_STRIDE + k0 + lr]        = acc0[r] * 0.125f;
            sc[(16 + r + 8 * hi) * SC_STRIDE + k0 + lr]   = acc1[r] * 0.125f;
        }
    }
    __syncthreads();

    // ---- phase 2: softmax per row; attn -> global once, P back to LDS ---
#pragma unroll
    for (int j = 0; j < 4; ++j) {
        const int m = wid * 4 + j;
        float* rowp = sc + (size_t)m * SC_STRIDE;
        float v[64];
        float mx = -3.0e38f;
#pragma unroll
        for (int i = 0; i < 64; ++i) { v[i] = rowp[i * 32 + lane]; mx = fmaxf(mx, v[i]); }
#pragma unroll
        for (int off = 16; off > 0; off >>= 1) mx = fmaxf(mx, __shfl_xor(mx, off, 32));
        float sum = 0.f;
#pragma unroll
        for (int i = 0; i < 64; ++i) { v[i] = __expf(v[i] - mx); sum += v[i]; }
#pragma unroll
        for (int off = 16; off > 0; off >>= 1) sum += __shfl_xor(sum, off, 32);
        const float inv = 1.0f / sum;
        float* gout = attn_out + ((size_t)bh * S_LEN + q0 + m) * S_LEN;
#pragma unroll
        for (int i = 0; i < 64; ++i) {
            const float p = v[i] * inv;
            rowp[i * 32 + lane] = p;
            gout[i * 32 + lane] = p;
        }
    }
    __syncthreads();

    // ---- phase 3: partial P*V over this wave's 256-wide k slice ---------
    v8f acc[2][4] = {};
    const float* prow0 = sc + (size_t)lr * SC_STRIDE + kbase + hi * 16;
    const float* prow1 = prow0 + (size_t)16 * SC_STRIDE;
#pragma unroll 2
    for (int kc = 0; kc < 256; kc += 32) {
        v16bf af0 = fragA_f32(prow0 + kc);   // ds_load_b128 + cvt_pk_bf16
        v16bf af1 = fragA_f32(prow1 + kc);
#pragma unroll
        for (int n = 0; n < 4; ++n) {
            const __bf16* vrow =
                vtb + ((size_t)bh * DKK + n * 16 + lr) * S_LEN + kbase + kc + hi * 16;
            __builtin_prefetch(vrow + 64, 0, 1);
            v16bf bf = fragB_bf16(vrow);
            acc[0][n] = WMMA_BF16(af0, bf, acc[0][n]);
            acc[1][n] = WMMA_BF16(af1, bf, acc[1][n]);
        }
    }
    __syncthreads();   // everyone done reading P; safe to reuse LDS

    // per-wave partial tiles into LDS: slice wid -> 32x64 floats
    float* part = sc + (size_t)wid * 2048;
#pragma unroll
    for (int t = 0; t < 2; ++t)
#pragma unroll
        for (int n = 0; n < 4; ++n)
#pragma unroll
            for (int r = 0; r < 8; ++r) {
                const int row = t * 16 + r + 8 * hi;
                const int col = n * 16 + lr;
                part[row * 64 + col] = acc[t][n][r];
            }
    __syncthreads();

    // deterministic cross-wave reduction + context writeback
    for (int i = tid; i < 2048; i += 256) {
        float s = 0.f;
#pragma unroll
        for (int w = 0; w < 8; ++w) s += sc[w * 2048 + i];
        const int row = i >> 6, d = i & 63;
        ctx[((size_t)(b * S_LEN + q0 + row)) * DM + h * DKK + d] = s;
    }
}

// -------------------------------------------------------------------------
extern "C" void kernel_launch(void* const* d_in, const int* in_sizes, int n_in,
                              void* d_out, int out_size, void* d_ws, size_t ws_size,
                              hipStream_t stream)
{
    const float* Q  = (const float*)d_in[0];
    const float* K  = (const float*)d_in[1];
    const float* V  = (const float*)d_in[2];
    const float* Wq = (const float*)d_in[3];
    const float* bq = (const float*)d_in[4];
    const float* Wk = (const float*)d_in[5];
    const float* bk = (const float*)d_in[6];
    const float* Wv = (const float*)d_in[7];
    const float* bv = (const float*)d_in[8];
    const float* Wo = (const float*)d_in[9];
    const float* bo = (const float*)d_in[10];

    float* outF  = (float*)d_out;                          // [2,2048,1024]
    float* attnF = outF + (size_t)2 * S_LEN * DM;          // [2,16,2048,2048]

    char* ws = (char*)d_ws;                                // 40 MB used
    __bf16* qb  = (__bf16*)(ws);                               // 8 MB
    __bf16* kb  = (__bf16*)(ws + (size_t)8  * 1024 * 1024);    // 8 MB
    __bf16* vtb = (__bf16*)(ws + (size_t)16 * 1024 * 1024);    // 8 MB
    float*  ctx = (float*) (ws + (size_t)24 * 1024 * 1024);    // 16 MB

    const dim3 blk(256);
    const size_t smem = (size_t)32 * SC_STRIDE * sizeof(float);   // 262,656 B

    proj_gemm_kernel<<<dim3(16, 16), blk, 0, stream>>>(Q, Wq, bq, qb,  nullptr, 0);
    proj_gemm_kernel<<<dim3(16, 16), blk, 0, stream>>>(K, Wk, bk, kb,  nullptr, 0);
    proj_gemm_kernel<<<dim3(16, 16), blk, 0, stream>>>(V, Wv, bv, vtb, nullptr, 1);
    fused_attn_kernel<<<dim3(64, 32), blk, smem, stream>>>(qb, kb, vtb, attnF, ctx);
    proj_gemm_kernel<<<dim3(16, 16), blk, 0, stream>>>(ctx, Wo, bo, nullptr, outF, 2);
}
